// point_cloud_decoder_30039001268461
// MI455X (gfx1250) — compile-verified, matrix operand order checked by reading, not run on hardware
//
#include <hip/hip_runtime.h>

typedef __attribute__((ext_vector_type(16))) _Float16 v16h;
typedef __attribute__((ext_vector_type(8)))  _Float16 v8h;
typedef __attribute__((ext_vector_type(8)))  float    v8f;

#define G 16
#define N 256
#define D 128
#define R 50
#define H 4
#define KNN 100
#define KPAD 112   // 7 * 16
#define RPAD 64    // 50 padded to 64 (2 WMMA K-steps)
#define NL 4
#define T 10
#define NODES (G * N)

__device__ __forceinline__ float gelu_tanh(float x) {
    const float c = 0.7978845608028654f;
    float t = tanhf(c * (x + 0.044715f * x * x * x));
    return 0.5f * x * (1.0f + t);
}

__device__ __forceinline__ v16h make_a_frag(const v8h* lo, const v8h* hi) {
    v8h a = *lo, b = *hi;
    return __builtin_shufflevector(a, b, 0, 1, 2, 3, 4, 5, 6, 7, 8, 9, 10, 11, 12, 13, 14, 15);
}

// -------------------- weight conversion + fragment swizzle --------------------
// dst layout: [kt][nt][lane][16]; element e of lane maps to
//   krow = kt*32 + (e<8 ? koff+e : 16+koff+(e-8)),  col = nt*16 + (lane&15),  koff = (lane>>4)*8
// rows >= srcK are zero-padded (used for the R=50 -> 64 edge weight).
__global__ void swizzle_w_kernel(const float* __restrict__ src, _Float16* __restrict__ dst,
                                 int Kp, int Nw, int srcK) {
    int i = blockIdx.x * blockDim.x + threadIdx.x;
    if (i >= Kp * Nw) return;
    int e = i & 15;
    int lane = (i >> 4) & 31;
    int tile = i >> 9;
    int ntiles = Nw >> 4;
    int kt = tile / ntiles, nt = tile - kt * ntiles;
    int r = lane & 15, koff = (lane >> 4) << 3;
    int krow = kt * 32 + ((e < 8) ? (koff + e) : (16 + koff + (e - 8)));
    int col = nt * 16 + r;
    dst[i] = (krow < srcK) ? (_Float16)src[krow * Nw + col] : (_Float16)0.f;
}

__global__ void copyf_kernel(const float* __restrict__ s, float* __restrict__ d, int n) {
    int i = blockIdx.x * blockDim.x + threadIdx.x;
    if (i < n) d[i] = s[i];
}

// -------------------- WMMA GEMM: Y[M,N] (+)= X[M,128] @ W[128,N] + bias --------------------
// Wsw is fragment-swizzled fp16. grid = (ceil(M/64), N/128), block = 256 (8 waves).
// Wave w owns 16-col tile (blockIdx.y*8 + w) and 4 M-tiles (64 rows) -> 16 WMMAs/wave.
// The 32KB weight slab for this block is staged into LDS with async global->LDS copies
// (CDNA5 GLOBAL_LOAD_ASYNC_TO_LDS_B128, tracked by ASYNCcnt), overlapped with the
// fp32->fp16 conversion of the X tile.
__global__ void gemm_wmma_k128(const float* __restrict__ X, const _Float16* __restrict__ Wsw,
                               const float* __restrict__ bias, float* __restrict__ Y,
                               int M, int Nw, int accumulate) {
    __shared__ _Float16 sx[64][D];        // 16 KB
    __shared__ _Float16 sw[4 * 8 * 512];  // 32 KB: [kt][nt_local][lane][16]
    int tid = threadIdx.x;
    int rowBase = blockIdx.x * 64;
    int ntiles = Nw >> 4;
    int nt0 = blockIdx.y * 8;

    // async-stage weight slab: 4 chunks (one per kt) of 8KB, 16B per lane per issue
    {
        unsigned ldsW = (unsigned)(uintptr_t)&sw[0];
        const char* gw = (const char*)Wsw;
#pragma unroll
        for (int kt = 0; kt < 4; ++kt) {
            size_t gbase = ((size_t)(kt * ntiles + nt0) * 512) * sizeof(_Float16);
#pragma unroll
            for (int it = 0; it < 2; ++it) {
                int off = (it * 256 + tid) * 16;
                unsigned long long ga = (unsigned long long)(uintptr_t)(gw + gbase + off);
                unsigned la = ldsW + kt * 8192 + off;
                asm volatile("global_load_async_to_lds_b128 %0, %1, off"
                             :: "v"(la), "v"(ga) : "memory");
            }
        }
    }

    // meanwhile: convert X tile to fp16 in LDS
    for (int i = tid; i < 64 * D; i += 256) {
        int r = i >> 7, c = i & 127;
        int gr = rowBase + r;
        sx[r][c] = (gr < M) ? (_Float16)X[gr * D + c] : (_Float16)0.f;
    }
    asm volatile("s_wait_asynccnt 0x0" ::: "memory");
    __syncthreads();

    int lane = tid & 31, wave = tid >> 5;
    int r = lane & 15;
    int koff = (lane >> 4) << 3;
    int nt = nt0 + wave;
    int col = nt * 16 + r;

    v8f acc[4];
#pragma unroll
    for (int mt = 0; mt < 4; ++mt) acc[mt] = (v8f){0.f, 0.f, 0.f, 0.f, 0.f, 0.f, 0.f, 0.f};

    const v16h* swv = (const v16h*)sw;
#pragma unroll
    for (int kt = 0; kt < 4; ++kt) {
        v16h b = swv[kt * 256 + wave * 32 + lane];
        int kb = kt * 32;
#pragma unroll
        for (int mt = 0; mt < 4; ++mt) {
            int arow = mt * 16 + r;
            v16h a = make_a_frag((const v8h*)&sx[arow][kb + koff],
                                 (const v8h*)&sx[arow][kb + 16 + koff]);
            acc[mt] = __builtin_amdgcn_wmma_f32_16x16x32_f16(false, a, false, b, (short)0,
                                                             acc[mt], false, false);
        }
    }

    float bv = bias[col];
    int rsub = (lane >> 4) << 3;
#pragma unroll
    for (int mt = 0; mt < 4; ++mt) {
        int rb = rowBase + mt * 16 + rsub;
#pragma unroll
        for (int i = 0; i < 8; ++i) {
            int gr = rb + i;
            if (gr < M) {
                float v = acc[mt][i] + bv;
                Y[gr * Nw + col] = accumulate ? (Y[gr * Nw + col] + v) : v;
            }
        }
    }
}

// -------------------- grid feature LN+gelu: Yup[G][D*27] -> gridf[G][27][D] --------------------
__global__ void grid_ln_gelu_kernel(const float* __restrict__ Yup, const float* __restrict__ gg,
                                    const float* __restrict__ bb, float* __restrict__ gridf) {
    __shared__ float sred[D];
    int gp = blockIdx.x;
    int g = gp / 27, pt = gp % 27;
    int t = threadIdx.x;
    float v = Yup[g * (D * 27) + t * 27 + pt];
    sred[t] = v; __syncthreads();
    for (int s = 64; s > 0; s >>= 1) { if (t < s) sred[t] += sred[t + s]; __syncthreads(); }
    float mean = sred[0] * (1.f / D); __syncthreads();
    float dv = v - mean;
    sred[t] = dv * dv; __syncthreads();
    for (int s = 64; s > 0; s >>= 1) { if (t < s) sred[t] += sred[t + s]; __syncthreads(); }
    float var = sred[0] * (1.f / D);
    float y = dv * rsqrtf(var + 1e-5f) * gg[t] + bb[t];
    gridf[(g * 27 + pt) * D + t] = gelu_tanh(y);
}

// -------------------- knn interpolate (k=3) + LN + gelu -> x --------------------
__global__ void knn_interp_kernel(const float* __restrict__ pos, const float* __restrict__ gridf,
                                  const float* __restrict__ gg, const float* __restrict__ bb,
                                  float* __restrict__ x) {
    __shared__ float sd[27];
    __shared__ int si[3];
    __shared__ float sw4[4];
    __shared__ float sred[D];
    int node = blockIdx.x;
    int g = node / N;
    int t = threadIdx.x;
    if (t < 27) {
        float gx = (float)(t / 9) - 1.f;
        float gy = (float)((t / 3) % 3) - 1.f;
        float gz = (float)(t % 3) - 1.f;
        float dx = pos[node * 3 + 0] - gx;
        float dy = pos[node * 3 + 1] - gy;
        float dz = pos[node * 3 + 2] - gz;
        sd[t] = dx * dx + dy * dy + dz * dz;
    }
    __syncthreads();
    if (t == 0) {
        float bv[3] = {1e30f, 1e30f, 1e30f};
        int bi[3] = {0, 0, 0};
        for (int j = 0; j < 27; ++j) {
            float v = sd[j];
            if (v < bv[0]) { bv[2]=bv[1]; bi[2]=bi[1]; bv[1]=bv[0]; bi[1]=bi[0]; bv[0]=v; bi[0]=j; }
            else if (v < bv[1]) { bv[2]=bv[1]; bi[2]=bi[1]; bv[1]=v; bi[1]=j; }
            else if (v < bv[2]) { bv[2]=v; bi[2]=j; }
        }
        float wsum = 0.f;
        for (int i = 0; i < 3; ++i) {
            float w = 1.f / (bv[i] + 1e-16f);
            sw4[i] = w; wsum += w; si[i] = bi[i];
        }
        sw4[3] = wsum;
    }
    __syncthreads();
    float v = (sw4[0] * gridf[(g * 27 + si[0]) * D + t] +
               sw4[1] * gridf[(g * 27 + si[1]) * D + t] +
               sw4[2] * gridf[(g * 27 + si[2]) * D + t]) / sw4[3];
    sred[t] = v; __syncthreads();
    for (int s = 64; s > 0; s >>= 1) { if (t < s) sred[t] += sred[t + s]; __syncthreads(); }
    float mean = sred[0] * (1.f / D); __syncthreads();
    float dv = v - mean;
    sred[t] = dv * dv; __syncthreads();
    for (int s = 64; s > 0; s >>= 1) { if (t < s) sred[t] += sred[t + s]; __syncthreads(); }
    float var = sred[0] * (1.f / D);
    float y = dv * rsqrtf(var + 1e-5f) * gg[t] + bb[t];
    x[node * D + t] = gelu_tanh(y);
}

// -------------------- radius graph (top-K=100 by distance) --------------------
__global__ void build_graph_kernel(const float* __restrict__ pos, int* __restrict__ idx,
                                   float* __restrict__ dist) {
    __shared__ float sd2[N];
    __shared__ float rv[N];
    __shared__ int ri[N];
    int node = blockIdx.x;
    int g = node / N, n = node % N;
    int t = threadIdx.x;
    float px = pos[node * 3 + 0], py = pos[node * 3 + 1], pz = pos[node * 3 + 2];
    int m = (g * N + t);
    float dx = px - pos[m * 3 + 0];
    float dy = py - pos[m * 3 + 1];
    float dz = pz - pos[m * 3 + 2];
    float d2 = dx * dx + dy * dy + dz * dz;
    if (t == n) d2 += 1e9f;
    sd2[t] = d2;
    __syncthreads();
    for (int k = 0; k < KNN; ++k) {
        rv[t] = sd2[t]; ri[t] = t; __syncthreads();
        for (int s = 128; s > 0; s >>= 1) {
            if (t < s && rv[t + s] < rv[t]) { rv[t] = rv[t + s]; ri[t] = ri[t + s]; }
            __syncthreads();
        }
        if (t == 0) {
            idx[node * KPAD + k] = ri[0];
            dist[node * KPAD + k] = sqrtf(rv[0] + 1e-12f);
            sd2[ri[0]] = 4e9f;
        }
        __syncthreads();
    }
    if (t >= KNN && t < KPAD) { idx[node * KPAD + t] = 0; dist[node * KPAD + t] = 1e9f; }
}

// -------------------- fused edge-embedding GEMM + transformer conv attention --------------------
// one block (256 thr, 8 waves) per node; Weh is fragment-swizzled [2][8][32][16]
__global__ void conv_attn_kernel(const float* __restrict__ xq, const float* __restrict__ xk,
                                 const float* __restrict__ xv, const int* __restrict__ idx,
                                 const float* __restrict__ dist, const _Float16* __restrict__ Weh,
                                 const float* __restrict__ eb, float* __restrict__ xout) {
    __shared__ _Float16 s_rbf[KPAD][RPAD];
    __shared__ _Float16 s_e[KPAD][D];
    __shared__ float s_q[D];
    __shared__ float s_logit[KPAD][H];
    __shared__ float s_alpha[KPAD][H];
    __shared__ int s_idx[KPAD];
    __shared__ float s_dist[KPAD];

    int node = blockIdx.x;
    int g = node / N;
    int tid = threadIdx.x;

    if (tid < KPAD) { s_idx[tid] = idx[node * KPAD + tid]; s_dist[tid] = dist[node * KPAD + tid]; }
    if (tid < D) s_q[tid] = xq[node * D + tid];
    __syncthreads();

    // rbf: coeff = -0.5/(CUTOFF/(R-1))^2 = -300.125 ; offs[i] = i * 2/49
    const float step = 2.0f / 49.0f;
    const float coeff = -300.125f;
    for (int i = tid; i < KPAD * RPAD; i += 256) {
        int j = i / RPAD, r = i % RPAD;
        float v = 0.f;
        if (r < R && j < KNN) {
            float d = s_dist[j] - step * (float)r;
            v = __expf(coeff * d * d);
        }
        s_rbf[j][r] = (_Float16)v;
    }
    __syncthreads();

    // e[112][128] = rbf[112][64] @ We[64][128] + eb ; wave -> 16-col tile, 7 M-tiles, 2 K-steps
    int lane = tid & 31, wave = tid >> 5;
    int r = lane & 15;
    int koff = (lane >> 4) << 3;
    int col = wave * 16 + r;
    float ebv = eb[col];
    const v16h* wv = (const v16h*)Weh;
#pragma unroll
    for (int kt = 0; kt < 2; ++kt) {
        v16h b = wv[(kt * 8 + wave) * 32 + lane];
        int kb = kt * 32;
#pragma unroll
        for (int mt = 0; mt < 7; ++mt) {
            int arow = mt * 16 + r;
            v16h a = make_a_frag((const v8h*)&s_rbf[arow][kb + koff],
                                 (const v8h*)&s_rbf[arow][kb + 16 + koff]);
            v8f acc = (v8f){0.f, 0.f, 0.f, 0.f, 0.f, 0.f, 0.f, 0.f};
            acc = __builtin_amdgcn_wmma_f32_16x16x32_f16(false, a, false, b, (short)0, acc, false, false);
            // accumulate the two K-steps via LDS (kt==0 writes, kt==1 adds)
            int rb = mt * 16 + ((lane >> 4) << 3);
            if (kt == 0) {
#pragma unroll
                for (int i = 0; i < 8; ++i) s_e[rb + i][col] = (_Float16)acc[i];
            } else {
#pragma unroll
                for (int i = 0; i < 8; ++i)
                    s_e[rb + i][col] = (_Float16)((float)s_e[rb + i][col] + acc[i] + ebv);
            }
        }
    }
    __syncthreads();

    // logits per (edge, head)
    if (tid < KPAD) {
        int j = tid;
        bool valid = (j < KNN) && (s_dist[j] <= 2.0f);
        float acch[H] = {0.f, 0.f, 0.f, 0.f};
        if (valid) {
            int base = (g * N + s_idx[j]) * D;
            for (int d = 0; d < D; ++d) {
                float kx = xk[base + d] + (float)s_e[j][d];
                acch[d >> 5] += s_q[d] * kx;
            }
        }
        const float scale = 0.17677669529663687f; // 1/sqrt(32)
        for (int h = 0; h < H; ++h) s_logit[j][h] = valid ? acch[h] * scale : -1e9f;
    }
    __syncthreads();

    // softmax over edges, per head; multiply by mask
    if (tid < H) {
        int h = tid;
        float mx = -1e30f;
        for (int j = 0; j < KPAD; ++j) mx = fmaxf(mx, s_logit[j][h]);
        float sum = 0.f;
        for (int j = 0; j < KPAD; ++j) sum += __expf(s_logit[j][h] - mx);
        float inv = 1.f / (sum + 1e-20f);
        for (int j = 0; j < KPAD; ++j) {
            bool valid = (j < KNN) && (s_dist[j] <= 2.0f);
            s_alpha[j][h] = valid ? __expf(s_logit[j][h] - mx) * inv : 0.f;
        }
    }
    __syncthreads();

    // aggregate
    if (tid < D) {
        int d = tid, h = d >> 5;
        float a = 0.f;
        for (int j = 0; j < KPAD; ++j) {
            float al = s_alpha[j][h];
            if (al != 0.f) {
                int base = (g * N + s_idx[j]) * D;
                a += al * (xv[base + d] + (float)s_e[j][d]);
            }
        }
        xout[node * D + d] = a;
    }
}

// -------------------- row LN + gelu --------------------
__global__ void ln_gelu_rows_kernel(const float* __restrict__ Y, const float* __restrict__ gg,
                                    const float* __restrict__ bb, float* __restrict__ out) {
    __shared__ float sred[D];
    int row = blockIdx.x, t = threadIdx.x;
    float v = Y[row * D + t];
    sred[t] = v; __syncthreads();
    for (int s = 64; s > 0; s >>= 1) { if (t < s) sred[t] += sred[t + s]; __syncthreads(); }
    float mean = sred[0] * (1.f / D); __syncthreads();
    float dv = v - mean;
    sred[t] = dv * dv; __syncthreads();
    for (int s = 64; s > 0; s >>= 1) { if (t < s) sred[t] += sred[t + s]; __syncthreads(); }
    float var = sred[0] * (1.f / D);
    float y = dv * rsqrtf(var + 1e-5f) * gg[t] + bb[t];
    out[row * D + t] = gelu_tanh(y);
}

// -------------------- position update: pos += x[:, :3]; x[:, 125:128] = pos --------------------
__global__ void pos_update_kernel(float* __restrict__ pos, float* __restrict__ x) {
    int n = blockIdx.x * blockDim.x + threadIdx.x;
    if (n < NODES) {
        for (int c = 0; c < 3; ++c) {
            float p = pos[n * 3 + c] + x[n * D + c];
            pos[n * 3 + c] = p;
            x[n * D + (D - 3) + c] = p;
        }
    }
}

// -------------------- output head: out[n] = [pos(3), x@Wout+b(10)] --------------------
__global__ void out_head_kernel(const float* __restrict__ pos, const float* __restrict__ x,
                                const float* __restrict__ ow, const float* __restrict__ ob,
                                float* __restrict__ out) {
    int n = blockIdx.x, t = threadIdx.x;
    if (t < 3) {
        out[n * (3 + T) + t] = pos[n * 3 + t];
    } else if (t < 3 + T) {
        int o = t - 3;
        float a = ob[o];
        for (int d = 0; d < D; ++d) a += x[n * D + d] * ow[d * T + o];
        out[n * (3 + T) + t] = a;
    }
}

// ==================== host side ====================
extern "C" void kernel_launch(void* const* d_in, const int* in_sizes, int n_in,
                              void* d_out, int out_size, void* d_ws, size_t ws_size,
                              hipStream_t stream) {
    (void)in_sizes; (void)n_in; (void)out_size; (void)ws_size;
    int p = 0;
    auto F = [&](int i) { return (const float*)d_in[i]; };
    const float* enc = F(p++);
    const float* posIn = F(p++);
    struct Lp {
        const float *eb, *ew, *kb, *kw, *qb, *qw, *sb, *sw, *vb, *vw;
        const float *f0b, *f0w, *f0lb, *f0lg, *f1b, *f1w, *f1lb, *f1lg;
    } L[NL];
    for (int l = 0; l < NL; ++l) {
        L[l].eb = F(p++); L[l].ew = F(p++);
        L[l].kb = F(p++); L[l].kw = F(p++);
        L[l].qb = F(p++); L[l].qw = F(p++);
        L[l].sb = F(p++); L[l].sw = F(p++);
        L[l].vb = F(p++); L[l].vw = F(p++);
        L[l].f0b = F(p++); L[l].f0w = F(p++); L[l].f0lb = F(p++); L[l].f0lg = F(p++);
        L[l].f1b = F(p++); L[l].f1w = F(p++); L[l].f1lb = F(p++); L[l].f1lg = F(p++);
    }
    const float* ln1b = F(p++); const float* ln1g = F(p++);
    const float* ln2b = F(p++); const float* ln2g = F(p++);
    const float* outb = F(p++); const float* outw = F(p++);
    const float* upb  = F(p++); const float* upw  = F(p++);

    char* base = (char*)d_ws;
    size_t off = 0;
    auto alloc = [&](size_t bytes) -> void* {
        off = (off + 255) & ~(size_t)255;
        void* r = base + off;
        off += bytes;
        return r;
    };
    _Float16* h_up = (_Float16*)alloc((size_t)D * (D * 27) * 2);
    _Float16 *h_q[NL], *h_k[NL], *h_v[NL], *h_s[NL], *h_f0[NL], *h_f1[NL], *h_e[NL];
    for (int l = 0; l < NL; ++l) {
        h_q[l] = (_Float16*)alloc(D * D * 2);
        h_k[l] = (_Float16*)alloc(D * D * 2);
        h_v[l] = (_Float16*)alloc(D * D * 2);
        h_s[l] = (_Float16*)alloc(D * D * 2);
        h_f0[l] = (_Float16*)alloc(D * D * 2);
        h_f1[l] = (_Float16*)alloc(D * D * 2);
        h_e[l] = (_Float16*)alloc(RPAD * D * 2);
    }
    float* Yup   = (float*)alloc((size_t)G * D * 27 * 4);
    float* gridf = (float*)alloc((size_t)G * 27 * D * 4);
    float* posc  = (float*)alloc((size_t)NODES * 3 * 4);
    float* bufX  = (float*)alloc((size_t)NODES * D * 4);
    float* bufY  = (float*)alloc((size_t)NODES * D * 4);
    float* bufQ  = (float*)alloc((size_t)NODES * D * 4);
    float* bufK  = (float*)alloc((size_t)NODES * D * 4);
    float* bufV  = (float*)alloc((size_t)NODES * D * 4);
    float* bufT  = (float*)alloc((size_t)NODES * D * 4);
    int*   idxb  = (int*)alloc((size_t)NODES * KPAD * 4);
    float* distb = (float*)alloc((size_t)NODES * KPAD * 4);

    // convert + fragment-swizzle weights to fp16
    swizzle_w_kernel<<<(D * D * 27 + 255) / 256, 256, 0, stream>>>(upw, h_up, D, D * 27, D);
    for (int l = 0; l < NL; ++l) {
        swizzle_w_kernel<<<(D * D + 255) / 256, 256, 0, stream>>>(L[l].qw, h_q[l], D, D, D);
        swizzle_w_kernel<<<(D * D + 255) / 256, 256, 0, stream>>>(L[l].kw, h_k[l], D, D, D);
        swizzle_w_kernel<<<(D * D + 255) / 256, 256, 0, stream>>>(L[l].vw, h_v[l], D, D, D);
        swizzle_w_kernel<<<(D * D + 255) / 256, 256, 0, stream>>>(L[l].sw, h_s[l], D, D, D);
        swizzle_w_kernel<<<(D * D + 255) / 256, 256, 0, stream>>>(L[l].f0w, h_f0[l], D, D, D);
        swizzle_w_kernel<<<(D * D + 255) / 256, 256, 0, stream>>>(L[l].f1w, h_f1[l], D, D, D);
        swizzle_w_kernel<<<(RPAD * D + 255) / 256, 256, 0, stream>>>(L[l].ew, h_e[l], RPAD, D, R);
    }
    copyf_kernel<<<(NODES * 3 + 255) / 256, 256, 0, stream>>>(posIn, posc, NODES * 3);

    // upscale -> Yup ; LN+gelu -> gridf ; knn interpolate -> bufX
    gemm_wmma_k128<<<dim3(1, 27), 256, 0, stream>>>(enc, h_up, upb, Yup, G, D * 27, 0);
    grid_ln_gelu_kernel<<<G * 27, D, 0, stream>>>(Yup, ln1g, ln1b, gridf);
    knn_interp_kernel<<<NODES, D, 0, stream>>>(posc, gridf, ln2g, ln2b, bufX);

    float* x = bufX;
    float* y = bufY;
    for (int l = 0; l < NL; ++l) {
        build_graph_kernel<<<NODES, N, 0, stream>>>(posc, idxb, distb);
        gemm_wmma_k128<<<dim3(NODES / 64, 1), 256, 0, stream>>>(x, h_q[l], L[l].qb, bufQ, NODES, D, 0);
        gemm_wmma_k128<<<dim3(NODES / 64, 1), 256, 0, stream>>>(x, h_k[l], L[l].kb, bufK, NODES, D, 0);
        gemm_wmma_k128<<<dim3(NODES / 64, 1), 256, 0, stream>>>(x, h_v[l], L[l].vb, bufV, NODES, D, 0);
        conv_attn_kernel<<<NODES, 256, 0, stream>>>(bufQ, bufK, bufV, idxb, distb, h_e[l], L[l].eb, y);
        gemm_wmma_k128<<<dim3(NODES / 64, 1), 256, 0, stream>>>(x, h_s[l], L[l].sb, y, NODES, D, 1);
        gemm_wmma_k128<<<dim3(NODES / 64, 1), 256, 0, stream>>>(y, h_f0[l], L[l].f0b, bufT, NODES, D, 0);
        ln_gelu_rows_kernel<<<NODES, D, 0, stream>>>(bufT, L[l].f0lg, L[l].f0lb, x);
        gemm_wmma_k128<<<dim3(NODES / 64, 1), 256, 0, stream>>>(x, h_f1[l], L[l].f1b, bufT, NODES, D, 0);
        ln_gelu_rows_kernel<<<NODES, D, 0, stream>>>(bufT, L[l].f1lg, L[l].f1lb, y);
        pos_update_kernel<<<(NODES + 255) / 256, 256, 0, stream>>>(posc, y);
        float* tmp = x; x = y; y = tmp;
    }

    out_head_kernel<<<NODES, 16, 0, stream>>>(posc, x, outw, outb, (float*)d_out);
}
